// MSCN_45801531245079
// MI455X (gfx1250) — matrix operand order repaired and checked by reading, not compiled, not run on hardware
//
#include <hip/hip_runtime.h>
#include <hip/hip_bf16.h>
#include <math.h>

// ---------------------------------------------------------------------------
// MSCN fused forward for MI455X (gfx1250, wave32, WMMA f16 -> f32 acc)
//   B=4096, LS=32, LP=16, LJ=16, SF=512, PF=64, JF=128, H=256
//
// v4: batch 8 B-fragment loads, then __builtin_amdgcn_sched_barrier(0), then
//     8 back-to-back v_wmma -- the fence stops the pre-RA scheduler from
//     sinking loads to their consumers, so load latency is pipelined across
//     the whole WMMA batch instead of exposed per-tile.
// ---------------------------------------------------------------------------

typedef __attribute__((ext_vector_type(16))) _Float16 v16h;
typedef __attribute__((ext_vector_type(8)))  _Float16 v8h;
typedef __attribute__((ext_vector_type(8)))  float    v8f;

#define HDIM     256
#define BK       32
#define ROWS     64
#define NTHREADS 256

__device__ __forceinline__ void sched_fence()
{
#if defined(__has_builtin)
#if __has_builtin(__builtin_amdgcn_sched_barrier)
    __builtin_amdgcn_sched_barrier(0);   // nothing may be reordered across
#endif
#endif
}

// ---------------------------------------------------------------------------
// One-time weight prep: Wt[n*K + k] = (f16) W[k*N + n]   (N == HDIM)
// ---------------------------------------------------------------------------
__global__ __launch_bounds__(NTHREADS)
void mscn_prep_kernel(const float* __restrict__ W, _Float16* __restrict__ Wt,
                      int K, int N)
{
    const int idx = blockIdx.x * NTHREADS + threadIdx.x;
    if (idx < K * N) {
        const int k = idx / N;
        const int n = idx % N;
        Wt[(size_t)n * K + k] = (_Float16)W[idx];
    }
}

// B-fragment: contiguous 32 bytes per lane from pre-transposed f16 weights
__device__ __forceinline__ v16h load_bfrag(const _Float16* __restrict__ Wt,
                                           int ldk, int k0, int n)
{
    return *(const v16h*)&Wt[(size_t)n * ldk + k0];
}

// A-fragment: two 16B halves from an f16 row (LDS or global)
__device__ __forceinline__ v16h load_afrag(const _Float16* __restrict__ p)
{
    const v8h* ap = (const v8h*)p;
    const v8h lo = ap[0];
    const v8h hi = ap[1];               // +8 halves (K+8)
    v16h a;
    #pragma unroll
    for (int i = 0; i < 8; ++i) { a[i] = lo[i]; a[8 + i] = hi[i]; }
    return a;
}

// ---------------------------------------------------------------------------
// Fused branch: X[B,L,F] -> Linear(F,H)+ReLU -> Linear(H,H)+ReLU
//               -> masked sum over L / len  -> hid[B, 3H] at col_off (f16)
// One block: 64 (b,l)-rows x 256 cols. 8 waves: 4 M-tiles x 2 N-halves.
// W1t: [HDIM][F] f16, W2t: [HDIM][HDIM] f16 (pre-transposed).
// ---------------------------------------------------------------------------
template <int F, int L>
__global__ __launch_bounds__(NTHREADS, 1)
void mscn_branch_kernel(const float*    __restrict__ X,
                        const int*      __restrict__ len,
                        const _Float16* __restrict__ W1t,
                        const float*    __restrict__ b1,
                        const _Float16* __restrict__ W2t,
                        const float*    __restrict__ b2,
                        _Float16*       __restrict__ hid,
                        int col_off)
{
    __shared__ _Float16 At[ROWS * BK];        // A tile (f16)          4 KB
    __shared__ _Float16 H1s[ROWS * HDIM];     // layer-1 activations  32 KB
    __shared__ float    Pool[4 * HDIM];       // per-batch pooled      4 KB
    __shared__ int      slen[4];

    const int tid    = threadIdx.x;
    const int lane   = tid & 31;
    const int w      = tid >> 5;
    const int mbase  = (w & 3) * 16;          // wave M offset inside block
    const int nbase0 = (w >> 2) * 128;        // wave N offset inside block
    const int row0   = blockIdx.x * ROWS;     // global row in [B*L]
    constexpr int NB = ROWS / L;              // batches covered by this block

    const int arow = lane & 15;               // A-fragment row within M tile
    const int kh   = lane >> 4;               // 0/1 lane half
    const int koff = kh * 16;                 // A-fragment K sub-offset
    const int nlan = lane & 15;               // B/C column within tile

    for (int i = tid; i < NB * HDIM; i += NTHREADS) Pool[i] = 0.0f;
    if (tid < NB) slen[tid] = len[row0 / L + tid];

    v8f acc[8] = {};

    // ---------------- layer 1: K over F ----------------
    for (int k0 = 0; k0 < F; k0 += BK) {
        __syncthreads();
        {   // stage A: rows row0..row0+63, cols k0..k0+31 (f32 -> f16)
            const int r = tid >> 2;
            const int c = (tid & 3) * 8;
            const float* src = X + (size_t)(row0 + r) * F + k0 + c;
            if (k0 + BK < F)
                __builtin_prefetch(src + BK, 0, 1);
            #pragma unroll
            for (int i = 0; i < 8; ++i)
                At[r * BK + c + i] = (_Float16)src[i];
        }
        __syncthreads();

        const v16h a = load_afrag(&At[(mbase + arow) * BK + koff]);

        v16h bf[8];
        #pragma unroll
        for (int t = 0; t < 8; ++t)
            bf[t] = load_bfrag(W1t, F, k0 + kh * 16, nbase0 + t * 16 + nlan);
        sched_fence();
        #pragma unroll
        for (int t = 0; t < 8; ++t)
            acc[t] = __builtin_amdgcn_wmma_f32_16x16x32_f16(
                         false, a, false, bf[t], (short)0, acc[t], false, false);
        sched_fence();
    }
    __syncthreads();

    // bias + ReLU -> H1s (f16)
    #pragma unroll
    for (int t = 0; t < 8; ++t) {
        const int   n  = nbase0 + t * 16 + nlan;
        const float bb = b1[n];
        #pragma unroll
        for (int r = 0; r < 8; ++r) {
            const int m = mbase + r + kh * 8;
            float v = acc[t][r] + bb;
            H1s[m * HDIM + n] = (_Float16)(v > 0.0f ? v : 0.0f);
        }
    }
    {   v8f z = {};
        #pragma unroll
        for (int t = 0; t < 8; ++t) acc[t] = z;
    }
    __syncthreads();

    // ---------------- layer 2: K over HDIM (A from LDS, no barriers) -------
    #pragma unroll
    for (int k0 = 0; k0 < HDIM; k0 += BK) {
        const v16h a = load_afrag(&H1s[(mbase + arow) * HDIM + k0 + koff]);

        v16h bf[8];
        #pragma unroll
        for (int t = 0; t < 8; ++t)
            bf[t] = load_bfrag(W2t, HDIM, k0 + kh * 16, nbase0 + t * 16 + nlan);
        sched_fence();
        #pragma unroll
        for (int t = 0; t < 8; ++t)
            acc[t] = __builtin_amdgcn_wmma_f32_16x16x32_f16(
                         false, a, false, bf[t], (short)0, acc[t], false, false);
        sched_fence();
    }

    // bias + ReLU + masked pooling (ds_add_f32 atomics into Pool)
    #pragma unroll
    for (int t = 0; t < 8; ++t) {
        const int   n  = nbase0 + t * 16 + nlan;
        const float bb = b2[n];
        #pragma unroll
        for (int r = 0; r < 8; ++r) {
            const int m  = mbase + r + kh * 8;
            const int bl = m / L;
            const int l  = m % L;
            float v = acc[t][r] + bb;
            v = v > 0.0f ? v : 0.0f;
            if (l < slen[bl]) atomicAdd(&Pool[bl * HDIM + n], v);
        }
    }
    __syncthreads();

    // pooled / len -> hid (f16), one column per thread per local batch
    for (int bl = 0; bl < NB; ++bl) {
        const int   b   = row0 / L + bl;
        const float inv = 1.0f / (float)slen[bl];
        hid[(size_t)b * (3 * HDIM) + col_off + tid] =
            (_Float16)(Pool[bl * HDIM + tid] * inv);
    }
}

// ---------------------------------------------------------------------------
// Head: hid[B,768] -> Linear(768,256)+ReLU -> Linear(256,1) -> sigmoid
// Wo1t: [HDIM][768] f16 (pre-transposed). No barriers in the K-loop.
// ---------------------------------------------------------------------------
__global__ __launch_bounds__(NTHREADS, 1)
void mscn_head_kernel(const _Float16* __restrict__ hid,
                      const _Float16* __restrict__ Wo1t,
                      const float*    __restrict__ bo1,
                      const float*    __restrict__ Wo2,
                      const float*    __restrict__ bo2,
                      float*          __restrict__ out)
{
    __shared__ _Float16 H1s[ROWS * HDIM];     // 32 KB
    __shared__ float    wvec[HDIM];           //  1 KB
    __shared__ float    red[NTHREADS];        //  1 KB

    const int tid    = threadIdx.x;
    const int lane   = tid & 31;
    const int w      = tid >> 5;
    const int mbase  = (w & 3) * 16;
    const int nbase0 = (w >> 2) * 128;
    const int row0   = blockIdx.x * ROWS;

    const int arow = lane & 15;
    const int kh   = lane >> 4;
    const int koff = kh * 16;
    const int nlan = lane & 15;

    wvec[tid] = Wo2[tid];

    v8f acc[8] = {};

    for (int k0 = 0; k0 < 3 * HDIM; k0 += BK) {
        const v16h a = load_afrag(
            &hid[(size_t)(row0 + mbase + arow) * (3 * HDIM) + k0 + koff]);

        v16h bf[8];
        #pragma unroll
        for (int t = 0; t < 8; ++t)
            bf[t] = load_bfrag(Wo1t, 3 * HDIM, k0 + kh * 16,
                               nbase0 + t * 16 + nlan);
        sched_fence();
        #pragma unroll
        for (int t = 0; t < 8; ++t)
            acc[t] = __builtin_amdgcn_wmma_f32_16x16x32_f16(
                         false, a, false, bf[t], (short)0, acc[t], false, false);
        sched_fence();
    }

    #pragma unroll
    for (int t = 0; t < 8; ++t) {
        const int   n  = nbase0 + t * 16 + nlan;
        const float bb = bo1[n];
        #pragma unroll
        for (int r = 0; r < 8; ++r) {
            const int m = mbase + r + kh * 8;
            float v = acc[t][r] + bb;
            H1s[m * HDIM + n] = (_Float16)(v > 0.0f ? v : 0.0f);
        }
    }
    __syncthreads();

    // parallel dot with Wo2: 4 threads per row, 64 cols each, LDS reduce
    {
        const int row = tid >> 2;
        const int seg = (tid & 3) * 64;
        float s = 0.0f;
        #pragma unroll 8
        for (int c = 0; c < 64; ++c)
            s += (float)H1s[row * HDIM + seg + c] * wvec[seg + c];
        red[tid] = s;
    }
    __syncthreads();
    if ((tid & 3) == 0) {
        const float tot = red[tid] + red[tid + 1] + red[tid + 2] + red[tid + 3]
                        + bo2[0];
        out[row0 + (tid >> 2)] = 1.0f / (1.0f + expf(-tot));
    }
}

// ---------------------------------------------------------------------------
extern "C" void kernel_launch(void* const* d_in, const int* in_sizes, int n_in,
                              void* d_out, int out_size, void* d_ws, size_t ws_size,
                              hipStream_t stream)
{
    const float* samples    = (const float*)d_in[0];
    const int*   sample_len = (const int*)  d_in[1];
    const float* predicates = (const float*)d_in[2];
    const int*   pred_len   = (const int*)  d_in[3];
    const float* joins      = (const float*)d_in[4];
    const int*   join_len   = (const int*)  d_in[5];
    // d_in[6] = flows (unused by the reference)
    const float* Ws1 = (const float*)d_in[7];
    const float* bs1 = (const float*)d_in[8];
    const float* Ws2 = (const float*)d_in[9];
    const float* bs2 = (const float*)d_in[10];
    const float* Wp1 = (const float*)d_in[11];
    const float* bp1 = (const float*)d_in[12];
    const float* Wp2 = (const float*)d_in[13];
    const float* bp2 = (const float*)d_in[14];
    const float* Wj1 = (const float*)d_in[15];
    const float* bj1 = (const float*)d_in[16];
    const float* Wj2 = (const float*)d_in[17];
    const float* bj2 = (const float*)d_in[18];
    const float* Wo1 = (const float*)d_in[19];
    const float* bo1 = (const float*)d_in[20];
    const float* Wo2 = (const float*)d_in[21];
    const float* bo2 = (const float*)d_in[22];

    float* out = (float*)d_out;

    constexpr int Bn = 4096;

    // workspace layout (f16): hid[4096*768], then transposed weights
    _Float16* hid  = (_Float16*)d_ws;
    _Float16* wp   = hid + (size_t)Bn * 768;
    _Float16* Ws1t = wp; wp += 512 * HDIM;
    _Float16* Ws2t = wp; wp += HDIM * HDIM;
    _Float16* Wp1t = wp; wp += 64 * HDIM;
    _Float16* Wp2t = wp; wp += HDIM * HDIM;
    _Float16* Wj1t = wp; wp += 128 * HDIM;
    _Float16* Wj2t = wp; wp += HDIM * HDIM;
    _Float16* Wo1t = wp; wp += 768 * HDIM;

    auto prep = [&](const float* W, _Float16* Wt, int K, int N) {
        const int total = K * N;
        mscn_prep_kernel<<<(total + NTHREADS - 1) / NTHREADS, NTHREADS, 0, stream>>>(
            W, Wt, K, N);
    };
    prep(Ws1, Ws1t, 512, HDIM);
    prep(Ws2, Ws2t, HDIM, HDIM);
    prep(Wp1, Wp1t, 64, HDIM);
    prep(Wp2, Wp2t, HDIM, HDIM);
    prep(Wj1, Wj1t, 128, HDIM);
    prep(Wj2, Wj2t, HDIM, HDIM);
    prep(Wo1, Wo1t, 768, HDIM);

    // samples: 4096*32 rows / 64 = 2048 blocks
    mscn_branch_kernel<512, 32><<<(Bn * 32) / ROWS, NTHREADS, 0, stream>>>(
        samples, sample_len, Ws1t, bs1, Ws2t, bs2, hid, 0);
    // predicates: 4096*16 rows / 64 = 1024 blocks
    mscn_branch_kernel<64, 16><<<(Bn * 16) / ROWS, NTHREADS, 0, stream>>>(
        predicates, pred_len, Wp1t, bp1, Wp2t, bp2, hid, 256);
    // joins
    mscn_branch_kernel<128, 16><<<(Bn * 16) / ROWS, NTHREADS, 0, stream>>>(
        joins, join_len, Wj1t, bj1, Wj2t, bj2, hid, 512);
    // head: 4096 rows / 64 = 64 blocks
    mscn_head_kernel<<<Bn / ROWS, NTHREADS, 0, stream>>>(
        hid, Wo1t, bo1, Wo2, bo2, out);
}
